// MambaFNO2d_25451976196546
// MI455X (gfx1250) — matrix operand order repaired and checked by reading, not compiled, not run on hardware
//
#include <hip/hip_runtime.h>
#include <hip/hip_bf16.h>
#include <math.h>

// ---------------------------------------------------------------------------
// MambaFNO2d forward for MI455X (gfx1250, wave32).
// WMMA f16 for conv/fc GEMMs, WMMA f32 for partial-DFT GEMMs,
// async global->LDS staging for all reused operands.
// B=8, C=64, H=W=256, L=4, MODES 20x20, fc1=128.
// ---------------------------------------------------------------------------

#define BB 8
#define CC 64
#define HH 256
#define WW 256
#define M1 20
#define M2 20
#define NLAYERS 4
#define FC1N 128
#define NROWS ((size_t)BB * CC * HH)      // 131072 "rows" of length W
#define NPAD 48                           // DFT N padded to 3 WMMA tiles

typedef __attribute__((ext_vector_type(16))) _Float16 v16h;
typedef __attribute__((ext_vector_type(8)))  float    v8f;
typedef __attribute__((ext_vector_type(2)))  float    v2f;

struct alignas(32) Frag {
    union {
        v16h v;
        uint4 q[2];
        _Float16 h[16];
    };
};
union F2 {
    v2f v;
    float2 f;
};

__device__ __forceinline__ float gelu_exact(float x) {
    return 0.5f * x * (1.0f + erff(x * 0.70710678118654752f));
}

// CDNA5 async global->LDS copy (16B per lane), tracked by ASYNCcnt.
__device__ __forceinline__ void async_copy_b128(void* lds_dst, const void* gsrc) {
    unsigned ldsa = (unsigned)(size_t)lds_dst;
    asm volatile("global_load_async_to_lds_b128 %0, %1, off"
                 :: "v"(ldsa), "v"(gsrc) : "memory");
}
__device__ __forceinline__ void async_wait0() {
    asm volatile("s_wait_asynccnt 0x0" ::: "memory");
}

// ---------------------------------------------------------------------------
// fc0 lift: x (B,H,W,3) -> hbuf f32 (B,C,H,W)  and  hf16 (B,H,W,C)
// ---------------------------------------------------------------------------
__global__ void fc0_kernel(const float* __restrict__ x,
                           const float* __restrict__ w,      // (3,64)
                           const float* __restrict__ bias,   // (64)
                           float* __restrict__ hbuf,
                           _Float16* __restrict__ hf16) {
    size_t pix = (size_t)blockIdx.x * blockDim.x + threadIdx.x;
    if (pix >= (size_t)BB * HH * WW) return;
    float x0 = x[pix * 3 + 0];
    float x1 = x[pix * 3 + 1];
    float x2 = x[pix * 3 + 2];
    size_t b  = pix >> 16;        // H*W = 65536
    size_t yx = pix & 65535u;
    for (int c = 0; c < CC; ++c) {
        float v = x0 * w[c] + x1 * w[64 + c] + x2 * w[128 + c] + bias[c];
        hbuf[((b * CC + c) << 16) + yx] = v;
        hf16[pix * CC + c] = (_Float16)v;
    }
}

// ---------------------------------------------------------------------------
// Twiddle tables (layer-invariant, built once):
//  t1t[n][w] (48x256): forward DFT over W, e^{-i 2pi kx w/256}/16, n=2kx+ri
//  t2t[w][n] (256x48): inverse DFT over W with hermitian fold:
//      T2[0]=1/16, T2[1]=0, T2[2kx]=2cos/16, T2[2kx+1]=-2sin/16
// ---------------------------------------------------------------------------
__global__ void build_tables_kernel(float* __restrict__ t1t,
                                    float* __restrict__ t2t) {
    int t = blockIdx.x * blockDim.x + threadIdx.x;
    if (t >= NPAD * 256) return;
    int w = t & 255;
    int n = t >> 8;                 // 0..47
    int kx = n >> 1, ri = n & 1;
    float th = 6.283185307179586f * (float)kx * (float)w * (1.0f / 256.0f);
    float s, c; __sincosf(th, &s, &c);
    float fwd = (n < 2 * M2) ? (((ri == 0) ? c : -s) * 0.0625f) : 0.0f;
    t1t[n * 256 + w] = fwd;
    float inv;
    if (n >= 2 * M2)   inv = 0.0f;
    else if (kx == 0)  inv = (ri == 0) ? 0.0625f : 0.0f;
    else               inv = ((ri == 0) ? 2.0f * c : -2.0f * s) * 0.0625f;
    t2t[w * NPAD + n] = inv;
}

// ---------------------------------------------------------------------------
// Combine convh + convw + pw (center tap) into one 3x3 f16 weight
// ---------------------------------------------------------------------------
__global__ void combine_w_kernel(const float* __restrict__ ch,
                                 const float* __restrict__ cw,
                                 const float* __restrict__ pw,
                                 const float* __restrict__ chb,
                                 const float* __restrict__ cwb,
                                 const float* __restrict__ pwb,
                                 int layer,
                                 _Float16* __restrict__ wcomb,
                                 float* __restrict__ biasc) {
    int t = blockIdx.x * blockDim.x + threadIdx.x;
    if (t >= 9 * CC * CC) return;
    int cin  = t & 63;
    int cout = (t >> 6) & 63;
    int tap  = t >> 12;
    int kh = tap / 3, kw = tap % 3;
    size_t wi = ((((size_t)layer * CC + cout) * CC + cin) * 3 + kh) * 3 + kw;
    float v = ch[wi] + cw[wi];
    if (kh == 1 && kw == 1) v += pw[((size_t)layer * CC + cout) * CC + cin];
    wcomb[((size_t)tap * CC + cout) * CC + cin] = (_Float16)v;
    if (t < CC) biasc[t] = chb[layer * CC + t] + cwb[layer * CC + t] + pwb[layer * CC + t];
}

// ---------------------------------------------------------------------------
// Implicit-GEMM 3x3 conv via v_wmma_f32_16x16x32_f16, weights in LDS.
// ---------------------------------------------------------------------------
__global__ __launch_bounds__(256)
void conv3x3_wmma_kernel(const _Float16* __restrict__ act,
                         const _Float16* __restrict__ wcomb,
                         float* __restrict__ out) {
    __shared__ _Float16 wlds[9 * CC * CC];          // 73728 B

    for (int i = threadIdx.x; i < (9 * CC * CC) / 8; i += 256)
        async_copy_b128(&wlds[i * 8], wcomb + (size_t)i * 8);
    async_wait0();
    __syncthreads();

    const int lane = threadIdx.x & 31;
    const int wave = threadIdx.x >> 5;
    const int x0 = wave * 32;
    const int y  = blockIdx.y;
    const int b  = blockIdx.z;
    const int m  = lane & 15;
    const int hi = lane >> 4;

    v8f acc[2][4] = {};

    for (int kh = 0; kh < 3; ++kh) {
        int ys = y + kh - 1;
        if (ys < 0 || ys >= HH) continue;           // wave-uniform skip
        const _Float16* actrow = act + (((size_t)b * HH + ys) << 14);
        __builtin_prefetch(actrow + ((size_t)x0 << 6) + (lane << 7), 0, 1);
        for (int kw = 0; kw < 3; ++kw) {
            int tap = kh * 3 + kw;
            const _Float16* wtap = wlds + tap * CC * CC;
            #pragma unroll
            for (int kc = 0; kc < 2; ++kc) {
                Frag a[2];
                int kb = kc * 32 + hi * 8;
                #pragma unroll
                for (int st = 0; st < 2; ++st) {
                    int xs = x0 + st * 16 + m + kw - 1;
                    bool inb = (xs >= 0) && (xs < WW);
                    const _Float16* ap = actrow + (((size_t)(inb ? xs : 0)) << 6) + kb;
                    if (inb) {
                        a[st].q[0] = *(const uint4*)(ap);
                        a[st].q[1] = *(const uint4*)(ap + 16);
                    } else {
                        a[st].q[0] = uint4{0u, 0u, 0u, 0u};
                        a[st].q[1] = uint4{0u, 0u, 0u, 0u};
                    }
                }
                #pragma unroll
                for (int nt = 0; nt < 4; ++nt) {
                    Frag bf;
                    const _Float16* wp = wtap + (nt * 16 + m) * CC + kc * 32 + hi * 8;
                    bf.q[0] = *(const uint4*)(wp);
                    bf.q[1] = *(const uint4*)(wp + 16);
                    acc[0][nt] = __builtin_amdgcn_wmma_f32_16x16x32_f16(
                        false, a[0].v, false, bf.v, (short)0, acc[0][nt], false, false);
                    acc[1][nt] = __builtin_amdgcn_wmma_f32_16x16x32_f16(
                        false, a[1].v, false, bf.v, (short)0, acc[1][nt], false, false);
                }
            }
        }
    }

    #pragma unroll
    for (int st = 0; st < 2; ++st) {
        #pragma unroll
        for (int nt = 0; nt < 4; ++nt) {
            int cout = nt * 16 + m;
            float* op = out + ((((size_t)b * CC + cout) * HH + y) << 8)
                            + x0 + st * 16 + hi * 8;
            float4 lo = make_float4(acc[st][nt][0], acc[st][nt][1],
                                    acc[st][nt][2], acc[st][nt][3]);
            float4 hi4 = make_float4(acc[st][nt][4], acc[st][nt][5],
                                     acc[st][nt][6], acc[st][nt][7]);
            *(float4*)(op)     = lo;
            *(float4*)(op + 4) = hi4;
        }
    }
}

// ---------------------------------------------------------------------------
// Forward partial DFT over W as f32 WMMA GEMM:
//   S1(131072 x 40) = h(131072 x 256) x T1(256 x 48-padded)
// Wave = 16-row tile; K in steps of 4 (V_WMMA_F32_16X16X4_F32); table in LDS.
// ---------------------------------------------------------------------------
__global__ __launch_bounds__(256)
void dft_w_wmma_kernel(const float* __restrict__ h,
                       const float* __restrict__ t1t,   // [48][256]
                       float* __restrict__ s1) {        // [row][40]
    __shared__ float tl[NPAD * 256];                    // 49152 B
    for (int i = threadIdx.x; i < (NPAD * 256) / 4; i += 256)
        async_copy_b128(&tl[i * 4], t1t + (size_t)i * 4);
    async_wait0();
    __syncthreads();

    const int lane = threadIdx.x & 31;
    const int wave = threadIdx.x >> 5;
    const int m  = lane & 15;
    const int hi = lane >> 4;
    const size_t row0 = ((size_t)blockIdx.x * 8 + wave) * 16;
    const float* arow = h + (row0 + m) * 256 + 2 * hi;

    v8f acc[3] = {};
    for (int k0 = 0; k0 < 256; k0 += 4) {
        F2 a; a.f = *(const float2*)(arow + k0);
        #pragma unroll
        for (int nt = 0; nt < 3; ++nt) {
            F2 b; b.f = *(const float2*)(&tl[(nt * 16 + m) * 256 + k0 + 2 * hi]);
            acc[nt] = __builtin_amdgcn_wmma_f32_16x16x4_f32(
                false, a.v, false, b.v, (short)0, acc[nt], false, false);
        }
    }

    // C/D: VGPR r, lane -> (M = r + hi*8, N = nt*16 + m)
    #pragma unroll
    for (int nt = 0; nt < 3; ++nt) {
        int n = nt * 16 + m;
        if (n < 2 * M2) {
            #pragma unroll
            for (int r = 0; r < 8; ++r)
                s1[(row0 + r + hi * 8) * (2 * M2) + n] = acc[nt][r];
        }
    }
}

// ---------------------------------------------------------------------------
// Forward partial DFT over H: s1 -> s2 (B,C,40,M2) complex
// ---------------------------------------------------------------------------
__global__ void dft_h_fwd_kernel(const float* __restrict__ s1,
                                 float* __restrict__ s2) {
    size_t t = (size_t)blockIdx.x * blockDim.x + threadIdx.x;
    if (t >= (size_t)BB * CC * 40 * M2) return;
    int kx = (int)(t % M2);
    int kyIdx = (int)((t / M2) % 40);
    size_t bc = t / (40 * M2);
    int ky = (kyIdx < M1) ? kyIdx : (kyIdx + 216);
    float th = -6.283185307179586f * (float)ky * (1.0f / 256.0f);
    float ss, cs; __sincosf(th, &ss, &cs);
    float cw = 1.0f, sw = 0.0f, re = 0.0f, im = 0.0f;
    const float* base = s1 + ((bc * HH) * M2 + kx) * 2;
    for (int y = 0; y < HH; ++y) {
        float vr = base[(size_t)y * M2 * 2];
        float vi = base[(size_t)y * M2 * 2 + 1];
        re += vr * cw - vi * sw;
        im += vr * sw + vi * cw;
        float cn = cw * cs - sw * ss;
        sw = sw * cs + cw * ss; cw = cn;
    }
    s2[t * 2]     = re * 0.0625f;
    s2[t * 2 + 1] = im * 0.0625f;
}

// ---------------------------------------------------------------------------
// Mode mixing: out[b,o,ky,kx] = sum_i in[b,i,ky,kx] * w(ky,kx)[i,o]  (complex)
// ---------------------------------------------------------------------------
__global__ void mode_mix_kernel(const float* __restrict__ s2,
                                const float* __restrict__ w1,
                                const float* __restrict__ w2,
                                int layer,
                                float* __restrict__ s3) {
    size_t t = (size_t)blockIdx.x * blockDim.x + threadIdx.x;
    if (t >= (size_t)BB * CC * 40 * M2) return;
    int kx = (int)(t % M2);
    int kyIdx = (int)((t / M2) % 40);
    int o = (int)((t / (40 * M2)) % CC);
    int b = (int)(t / ((size_t)40 * M2 * CC));
    const float* wsel = (kyIdx < M1) ? w1 : w2;
    int kyw = (kyIdx < M1) ? kyIdx : (kyIdx - M1);
    float re = 0.0f, im = 0.0f;
    for (int i = 0; i < CC; ++i) {
        size_t si = (((size_t)(b * CC + i)) * (40 * M2) + (size_t)kyIdx * M2 + kx) * 2;
        float ar = s2[si], ai = s2[si + 1];
        size_t wi = (((((size_t)layer * CC + i) * CC + o) * M1 + kyw) * M2 + kx) * 2;
        float br = wsel[wi], bi = wsel[wi + 1];
        re += ar * br - ai * bi;
        im += ar * bi + ai * br;
    }
    size_t oi = (((size_t)(b * CC + o)) * (40 * M2) + (size_t)kyIdx * M2 + kx) * 2;
    s3[oi] = re; s3[oi + 1] = im;
}

// ---------------------------------------------------------------------------
// Inverse partial DFT over H: s3 (B,C,40,M2) -> s4p (row, 48-padded), 1/16.
// Twiddles via 256-entry LDS cos/sin tables (angles are mod 256).
// ---------------------------------------------------------------------------
__global__ __launch_bounds__(256)
void idft_h_kernel(const float* __restrict__ s3,
                   float* __restrict__ s4p) {
    __shared__ float ct[256], st[256];
    {
        int i = threadIdx.x;
        float s, c;
        __sincosf(6.283185307179586f * (float)i * (1.0f / 256.0f), &s, &c);
        ct[i] = c; st[i] = s;
    }
    __syncthreads();

    size_t t = (size_t)blockIdx.x * blockDim.x + threadIdx.x;
    if (t >= (size_t)BB * CC * HH * M2) return;
    int kx = (int)(t % M2);
    int y  = (int)((t / M2) % HH);
    size_t bc = t / ((size_t)M2 * HH);
    size_t row = bc * HH + y;
    float re = 0.0f, im = 0.0f;
    const float* base = s3 + bc * (40 * M2) * 2;
    #pragma unroll 4
    for (int kyIdx = 0; kyIdx < 40; ++kyIdx) {
        int ky = (kyIdx < M1) ? kyIdx : (kyIdx + 216);
        int ang = (ky * y) & 255;
        float c = ct[ang], s = st[ang];
        float vr = base[((size_t)kyIdx * M2 + kx) * 2];
        float vi = base[((size_t)kyIdx * M2 + kx) * 2 + 1];
        re += vr * c - vi * s;
        im += vr * s + vi * c;
    }
    s4p[row * NPAD + 2 * kx]     = re * 0.0625f;
    s4p[row * NPAD + 2 * kx + 1] = im * 0.0625f;
    if (kx < 4) {                       // zero the N padding 40..47
        s4p[row * NPAD + 40 + 2 * kx]     = 0.0f;
        s4p[row * NPAD + 40 + 2 * kx + 1] = 0.0f;
    }
}

// ---------------------------------------------------------------------------
// Fused inverse DFT over W (f32 WMMA GEMM) + residual + bias + GELU + writeout:
//   spec(131072 x 256) = s4p(131072 x 48) x T2(48 x 256)
// Wave = 16 rows x 64 cols; 8 waves/block cover 2 row-tiles.
// ---------------------------------------------------------------------------
__global__ __launch_bounds__(256)
void idft_w_fuse_wmma_kernel(const float* __restrict__ s4p,  // [row][48]
                             const float* __restrict__ t2t,  // [256][48]
                             const float* __restrict__ conv,
                             const float* __restrict__ biasc,
                             int apply_gelu,
                             float* __restrict__ hbuf,
                             _Float16* __restrict__ hf16) {
    __shared__ float tl[256 * NPAD];                 // 49152 B
    for (int i = threadIdx.x; i < (256 * NPAD) / 4; i += 256)
        async_copy_b128(&tl[i * 4], t2t + (size_t)i * 4);
    async_wait0();
    __syncthreads();

    const int lane = threadIdx.x & 31;
    const int wave = threadIdx.x >> 5;
    const int m  = lane & 15;
    const int hi = lane >> 4;
    const size_t row0 = ((size_t)blockIdx.x * 2 + (wave >> 2)) * 16;
    const int cgrp = (wave & 3) * 64;
    const float* arow = s4p + (row0 + m) * NPAD + 2 * hi;

    v8f acc[4] = {};
    #pragma unroll
    for (int k0 = 0; k0 < NPAD; k0 += 4) {
        F2 a; a.f = *(const float2*)(arow + k0);
        #pragma unroll
        for (int nt = 0; nt < 4; ++nt) {
            F2 b; b.f = *(const float2*)(&tl[(cgrp + nt * 16 + m) * NPAD + k0 + 2 * hi]);
            acc[nt] = __builtin_amdgcn_wmma_f32_16x16x4_f32(
                false, a.v, false, b.v, (short)0, acc[nt], false, false);
        }
    }

    #pragma unroll
    for (int nt = 0; nt < 4; ++nt) {
        int col = cgrp + nt * 16 + m;
        #pragma unroll
        for (int r = 0; r < 8; ++r) {
            size_t row = row0 + r + hi * 8;
            size_t idx = (row << 8) + col;
            int c = (int)((row >> 8) & 63);
            float v = conv[idx] + acc[nt][r] + biasc[c];
            if (apply_gelu) v = gelu_exact(v);
            hbuf[idx] = v;
            size_t b = row >> 14;
            size_t pix = (b << 16) + ((row & 255) << 8) + col;
            hf16[pix * CC + c] = (_Float16)v;
        }
    }
}

// ---------------------------------------------------------------------------
// fc1 weight prep: (64,128) f32 -> (128,64) f16
// ---------------------------------------------------------------------------
__global__ void prep_fc1_kernel(const float* __restrict__ w,
                                _Float16* __restrict__ out) {
    int t = blockIdx.x * blockDim.x + threadIdx.x;
    if (t >= FC1N * CC) return;
    int cin = t & 63;
    int cout = t >> 6;
    out[t] = (_Float16)w[cin * FC1N + cout];
}

// ---------------------------------------------------------------------------
// Fused head: fc1 (WMMA, weights in LDS) + bias + GELU + fc2 dot + bias.
// ---------------------------------------------------------------------------
__global__ __launch_bounds__(256)
void fc_head_wmma_kernel(const _Float16* __restrict__ act,   // (P,64)
                         const _Float16* __restrict__ fc1wt, // (128,64)
                         const float* __restrict__ fc1b,
                         const float* __restrict__ fc2w,
                         const float* __restrict__ fc2b,
                         float* __restrict__ out) {
    __shared__ _Float16 wlds[FC1N * CC];            // 16384 B
    for (int i = threadIdx.x; i < (FC1N * CC) / 8; i += 256)
        async_copy_b128(&wlds[i * 8], fc1wt + (size_t)i * 8);
    async_wait0();
    __syncthreads();

    const int lane = threadIdx.x & 31;
    const int wave = threadIdx.x >> 5;
    const size_t tile = (size_t)blockIdx.x * 8 + wave;
    const size_t pix0 = tile * 16;
    const int m  = lane & 15;
    const int hi = lane >> 4;
    const _Float16* arow = act + (pix0 + m) * CC;

    v8f acc[8] = {};
    #pragma unroll
    for (int kc = 0; kc < 2; ++kc) {
        Frag a;
        int kb = kc * 32 + hi * 8;
        a.q[0] = *(const uint4*)(arow + kb);
        a.q[1] = *(const uint4*)(arow + kb + 16);
        #pragma unroll
        for (int nt = 0; nt < 8; ++nt) {
            Frag bf;
            const _Float16* wp = wlds + (size_t)(nt * 16 + m) * CC + kc * 32 + hi * 8;
            bf.q[0] = *(const uint4*)(wp);
            bf.q[1] = *(const uint4*)(wp + 16);
            acc[nt] = __builtin_amdgcn_wmma_f32_16x16x32_f16(
                false, a.v, false, bf.v, (short)0, acc[nt], false, false);
        }
    }

    float partial[8];
    #pragma unroll
    for (int r = 0; r < 8; ++r) partial[r] = 0.0f;
    #pragma unroll
    for (int nt = 0; nt < 8; ++nt) {
        int n = nt * 16 + m;
        float bn = fc1b[n], w2 = fc2w[n];
        #pragma unroll
        for (int r = 0; r < 8; ++r) {
            float v = acc[nt][r] + bn;
            v = gelu_exact(v);
            partial[r] += v * w2;
        }
    }
    #pragma unroll
    for (int r = 0; r < 8; ++r) {
        float s = partial[r];
        s += __shfl_xor(s, 1, 32);
        s += __shfl_xor(s, 2, 32);
        s += __shfl_xor(s, 4, 32);
        s += __shfl_xor(s, 8, 32);
        partial[r] = s;
    }
    if (m == 0) {
        float b2 = fc2b[0];
        #pragma unroll
        for (int r = 0; r < 8; ++r)
            out[pix0 + (size_t)hi * 8 + r] = partial[r] + b2;
    }
}

// ---------------------------------------------------------------------------
// Workspace layout (bytes)
// ---------------------------------------------------------------------------
#define OFF_HBUF   ((size_t)0)
#define SZ_HBUF    ((size_t)BB * CC * HH * WW * 4)              // 134217728
#define OFF_CONV   (OFF_HBUF + SZ_HBUF)
#define SZ_CONV    SZ_HBUF
#define OFF_HF16   (OFF_CONV + SZ_CONV)
#define SZ_HF16    ((size_t)BB * CC * HH * WW * 2)              // 67108864
#define OFF_S1     (OFF_HF16 + SZ_HF16)
#define SZ_S1      (NROWS * 2 * M2 * 4)                         // 20971520
#define OFF_S4     (OFF_S1 + SZ_S1)
#define SZ_S4      (NROWS * NPAD * 4)                           // 25165824
#define OFF_S2     (OFF_S4 + SZ_S4)
#define SZ_S2      ((size_t)BB * CC * 40 * M2 * 2 * 4)          // 3276800
#define OFF_S3     (OFF_S2 + SZ_S2)
#define SZ_S3      SZ_S2
#define OFF_WCOMB  (OFF_S3 + SZ_S3)
#define SZ_WCOMB   ((size_t)9 * CC * CC * 2)                    // 73728
#define OFF_BIASC  (OFF_WCOMB + SZ_WCOMB)
#define SZ_BIASC   ((size_t)256)
#define OFF_FC1WT  (OFF_BIASC + SZ_BIASC)
#define SZ_FC1WT   ((size_t)FC1N * CC * 2)                      // 16384
#define OFF_T1T    (OFF_FC1WT + SZ_FC1WT)
#define SZ_T1T     ((size_t)NPAD * 256 * 4)                     // 49152
#define OFF_T2T    (OFF_T1T + SZ_T1T)
#define SZ_T2T     SZ_T1T

extern "C" void kernel_launch(void* const* d_in, const int* in_sizes, int n_in,
                              void* d_out, int out_size, void* d_ws, size_t ws_size,
                              hipStream_t stream) {
    (void)in_sizes; (void)n_in; (void)out_size; (void)ws_size;
    const float* x       = (const float*)d_in[0];
    const float* fc0_w   = (const float*)d_in[1];
    const float* fc0_b   = (const float*)d_in[2];
    const float* w1      = (const float*)d_in[3];
    const float* w2      = (const float*)d_in[4];
    const float* convh_w = (const float*)d_in[5];
    const float* convh_b = (const float*)d_in[6];
    const float* convw_w = (const float*)d_in[7];
    const float* convw_b = (const float*)d_in[8];
    const float* pw_w    = (const float*)d_in[9];
    const float* pw_b    = (const float*)d_in[10];
    const float* fc1_w   = (const float*)d_in[11];
    const float* fc1_b   = (const float*)d_in[12];
    const float* fc2_w   = (const float*)d_in[13];
    const float* fc2_b   = (const float*)d_in[14];

    char* ws = (char*)d_ws;
    float*     HBUF  = (float*)(ws + OFF_HBUF);
    float*     CONV  = (float*)(ws + OFF_CONV);
    _Float16*  HF16  = (_Float16*)(ws + OFF_HF16);
    float*     S1    = (float*)(ws + OFF_S1);
    float*     S2    = (float*)(ws + OFF_S2);
    float*     S3    = (float*)(ws + OFF_S3);
    float*     S4P   = (float*)(ws + OFF_S4);
    _Float16*  WCOMB = (_Float16*)(ws + OFF_WCOMB);
    float*     BIASC = (float*)(ws + OFF_BIASC);
    _Float16*  FC1WT = (_Float16*)(ws + OFF_FC1WT);
    float*     T1T   = (float*)(ws + OFF_T1T);
    float*     T2T   = (float*)(ws + OFF_T2T);

    const size_t NPIX = (size_t)BB * HH * WW;          // 524288

    fc0_kernel<<<(unsigned)((NPIX + 255) / 256), 256, 0, stream>>>(
        x, fc0_w, fc0_b, HBUF, HF16);
    prep_fc1_kernel<<<(FC1N * CC + 255) / 256, 256, 0, stream>>>(fc1_w, FC1WT);
    build_tables_kernel<<<(NPAD * 256 + 255) / 256, 256, 0, stream>>>(T1T, T2T);

    for (int l = 0; l < NLAYERS; ++l) {
        combine_w_kernel<<<(9 * CC * CC + 255) / 256, 256, 0, stream>>>(
            convh_w, convw_w, pw_w, convh_b, convw_b, pw_b, l, WCOMB, BIASC);
        conv3x3_wmma_kernel<<<dim3(1, HH, BB), 256, 0, stream>>>(
            HF16, WCOMB, CONV);
        dft_w_wmma_kernel<<<(unsigned)(NROWS / 16 / 8), 256, 0, stream>>>(
            HBUF, T1T, S1);
        dft_h_fwd_kernel<<<(unsigned)(((size_t)BB * CC * 40 * M2 + 255) / 256), 256, 0, stream>>>(
            S1, S2);
        mode_mix_kernel<<<(unsigned)(((size_t)BB * CC * 40 * M2 + 255) / 256), 256, 0, stream>>>(
            S2, w1, w2, l, S3);
        idft_h_kernel<<<(unsigned)(((size_t)BB * CC * HH * M2 + 255) / 256), 256, 0, stream>>>(
            S3, S4P);
        idft_w_fuse_wmma_kernel<<<(unsigned)(NROWS / 16 / 2), 256, 0, stream>>>(
            S4P, T2T, CONV, BIASC, (l < NLAYERS - 1) ? 1 : 0, HBUF, HF16);
    }

    fc_head_wmma_kernel<<<(unsigned)(NPIX / 16 / 8), 256, 0, stream>>>(
        HF16, FC1WT, fc1_b, fc2_w, fc2_b, (float*)d_out);
}
